// EigenGPT2_721554505772
// MI455X (gfx1250) — matrix operand order repaired and checked by reading, not compiled, not run on hardware
//
#include <hip/hip_runtime.h>
#include <math.h>

// ---------------- model dims ----------------
#define Bq 2
#define Sq 1024
#define Dm 1024
#define Hh 16
#define HDm 64
#define Lq 4
#define Kq 64
#define Vq 50257
#define Mrows (Bq * Sq)

typedef unsigned short bf16_t;

// ---------------- WMMA types ----------------
typedef __attribute__((ext_vector_type(16))) __bf16 v16bf;
typedef __attribute__((ext_vector_type(8)))  float  v8f;

union FragBF { v16bf v; uint4 q[2]; };
union U8     { uint4 q; unsigned short h[8]; };

__device__ __forceinline__ unsigned short f2bf(float x) {
  unsigned int u = __float_as_uint(x);
  u += 0x7fffu + ((u >> 16) & 1u);          // round-to-nearest-even
  return (unsigned short)(u >> 16);
}
__device__ __forceinline__ float bf2f(unsigned short h) {
  return __uint_as_float(((unsigned int)h) << 16);
}

__device__ __forceinline__ v8f wmma_bf16(v16bf a, v16bf b, v8f c) {
  return __builtin_amdgcn_wmma_f32_16x16x32_bf16(
      false, a, false, b, (short)0, c, false, false);
}

// gfx1250 async global->LDS copy (ASYNCcnt-tracked), 16B per lane
__device__ __forceinline__ void async_ld_b128(unsigned ldsoff, const void* g) {
  asm volatile("global_load_async_to_lds_b128 %0, %1, off"
               :: "v"(ldsoff), "v"(g) : "memory");
}
__device__ __forceinline__ void wait_async() {
  asm volatile("s_wait_asynccnt 0x0" ::: "memory");
}

// ---------------- f32 -> bf16 bulk convert ----------------
__global__ __launch_bounds__(256) void k_cvtbf(
    const float* __restrict__ x, bf16_t* __restrict__ y, long n) {
  long stride = (long)gridDim.x * 256 * 8;
  for (long i = ((long)blockIdx.x * 256 + threadIdx.x) * 8; i < n; i += stride) {
    float4 f0 = *(const float4*)(x + i);
    float4 f1 = *(const float4*)(x + i + 4);
    U8 u;
    u.h[0] = f2bf(f0.x); u.h[1] = f2bf(f0.y); u.h[2] = f2bf(f0.z); u.h[3] = f2bf(f0.w);
    u.h[4] = f2bf(f1.x); u.h[5] = f2bf(f1.y); u.h[6] = f2bf(f1.z); u.h[7] = f2bf(f1.w);
    *(uint4*)(y + i) = u.q;
  }
}

// ---------------- embedding (f32 residual stream) ----------------
__global__ __launch_bounds__(256) void k_embed(
    const int* __restrict__ ids, const float* __restrict__ wte,
    const float* __restrict__ wpe, float* __restrict__ x) {
  int row = blockIdx.x;
  int s = row % Sq;
  long id = ids[row];
  for (int c = threadIdx.x; c < Dm; c += 256)
    x[(size_t)row * Dm + c] = wte[(size_t)id * Dm + c] + wpe[(size_t)s * Dm + c];
}

// ---------------- layernorm: f32 in, bf16 out ----------------
__global__ __launch_bounds__(256) void k_ln(
    const float* __restrict__ x, const float* __restrict__ w,
    const float* __restrict__ b, bf16_t* __restrict__ y) {
  __shared__ float rs_[256];
  __shared__ float rq_[256];
  int row = blockIdx.x, tid = threadIdx.x;
  const float* xr = x + (size_t)row * Dm;
  float s = 0.f, q = 0.f;
  for (int c = tid; c < Dm; c += 256) { float v = xr[c]; s += v; q += v * v; }
  rs_[tid] = s; rq_[tid] = q;
  __syncthreads();
  for (int st = 128; st > 0; st >>= 1) {
    if (tid < st) { rs_[tid] += rs_[tid + st]; rq_[tid] += rq_[tid + st]; }
    __syncthreads();
  }
  float mean = rs_[0] * (1.f / Dm);
  float var  = rq_[0] * (1.f / Dm) - mean * mean;
  float rstd = rsqrtf(var + 1e-5f);
  for (int c = tid; c < Dm; c += 256)
    y[(size_t)row * Dm + c] = f2bf((xr[c] - mean) * rstd * w[c] + b[c]);
}

// ---------------- GEMM: bf16 x bf16 -> f32 acc, double-buffered LDS ------
// A: bf16 [M,Kd] (lda), optional f32 mean subtract along Kd (manual path).
// W: transb==0 -> bf16 [N,Kd] (C = A W^T, async-staged when no N-edge);
//    transb==1 -> bf16 [Kd,N] (C = A W, manual transpose staging).
// Outputs: Cf (f32, optional) and/or Cbf (bf16, optional), same ldc.
#define BM 128
#define BN 128
#define BK 32
#define LSTR 40   // LDS row stride in halves (16B-aligned rows)

__global__ __launch_bounds__(256) void k_gemm(
    const bf16_t* __restrict__ A, int lda,
    const bf16_t* __restrict__ W,
    const float* __restrict__ bias,
    const float* __restrict__ resid,
    const float* __restrict__ amean,
    float* __restrict__ Cf, bf16_t* __restrict__ Cbf, int ldc,
    int M, int N, int Kd, int transb, int gelu) {
  __shared__ __attribute__((aligned(16))) unsigned short As[2][BM * LSTR];
  __shared__ __attribute__((aligned(16))) unsigned short Bs[2][BN * LSTR];
  unsigned asb = (unsigned)(size_t)(void*)As;
  unsigned bsb = (unsigned)(size_t)(void*)Bs;

  int tid  = threadIdx.x;
  int lane = tid & 31;
  int wv   = tid >> 5;         // 0..7
  int wm   = wv >> 1;          // 0..3 : 32-row strip
  int wn   = wv & 1;           // 0..1 : 64-col strip
  int m0 = blockIdx.y * BM, n0 = blockIdx.x * BN;
  bool edge = (n0 + BN > N);   // uniform across block

  v8f acc[2][4];
#pragma unroll
  for (int i = 0; i < 2; ++i)
#pragma unroll
    for (int j = 0; j < 4; ++j)
#pragma unroll
      for (int r = 0; r < 8; ++r) acc[i][j][r] = 0.f;

  int m16 = lane & 15;
  int kb8 = (lane >> 4) << 3;   // A-frag K base (0/8)
  int b16 = (lane >> 4) << 4;   // unused (0/16); B-frag also uses kb8

  // stage tile at k0 into buffer bufi (async fast path / manual slow paths)
  auto stage = [&](int k0, int bufi) {
    unsigned ab_ = asb + (unsigned)bufi * (BM * LSTR * 2);
    unsigned bb_ = bsb + (unsigned)bufi * (BN * LSTR * 2);
    if (!amean) {
#pragma unroll
      for (int i = 0; i < 2; ++i) {
        int idx = tid + i * 256;
        int r = idx >> 2, c = idx & 3;
        async_ld_b128(ab_ + r * (LSTR * 2) + c * 16,
                      A + (size_t)(m0 + r) * lda + k0 + c * 8);
      }
    } else {
#pragma unroll
      for (int i = 0; i < 2; ++i) {
        int idx = tid + i * 256;
        int r = idx >> 2, c = idx & 3;
        U8 u; u.q = *(const uint4*)(A + (size_t)(m0 + r) * lda + k0 + c * 8);
        const float* mm = amean + k0 + c * 8;
#pragma unroll
        for (int e = 0; e < 8; ++e) u.h[e] = f2bf(bf2f(u.h[e]) - mm[e]);
        *(uint4*)&As[bufi][r * LSTR + c * 8] = u.q;
      }
    }
    if (!transb && !edge) {
#pragma unroll
      for (int i = 0; i < 2; ++i) {
        int idx = tid + i * 256;
        int n = idx >> 2, c = idx & 3;
        async_ld_b128(bb_ + n * (LSTR * 2) + c * 16,
                      W + (size_t)(n0 + n) * Kd + k0 + c * 8);
      }
    } else if (!transb) {
#pragma unroll
      for (int i = 0; i < 2; ++i) {
        int idx = tid + i * 256;
        int n = idx >> 2, c = idx & 3;
        int gn = n0 + n;
        uint4 z = make_uint4(0u, 0u, 0u, 0u);
        if (gn < N) z = *(const uint4*)(W + (size_t)gn * Kd + k0 + c * 8);
        *(uint4*)&Bs[bufi][n * LSTR + c * 8] = z;
      }
    } else {
#pragma unroll
      for (int i = 0; i < 16; ++i) {
        int idx = tid + i * 256;            // 4096 scalars
        int kk = idx >> 7, n = idx & 127;
        int gn = n0 + n;
        bf16_t f = 0;
        if (gn < N) f = W[(size_t)(k0 + kk) * N + gn];
        Bs[bufi][n * LSTR + kk] = f;
      }
    }
  };

  int nk = Kd / BK;
  stage(0, 0);
  for (int kt = 0; kt < nk; ++kt) {
    wait_async();           // this wave's tile-kt async copies landed
    __syncthreads();        // all waves done writing tile kt / reading kt-1
    if (kt + 1 < nk) stage((kt + 1) * BK, (kt + 1) & 1);
    int bi = kt & 1;

    FragBF a[2], bb[4];
#pragma unroll
    for (int ms = 0; ms < 2; ++ms) {
      const unsigned short* ap = &As[bi][(wm * 32 + ms * 16 + m16) * LSTR];
      a[ms].q[0] = *(const uint4*)(ap + kb8);
      a[ms].q[1] = *(const uint4*)(ap + 16 + kb8);
    }
#pragma unroll
    for (int ns = 0; ns < 4; ++ns) {
      const unsigned short* bp = &Bs[bi][(wn * 64 + ns * 16 + m16) * LSTR];
      bb[ns].q[0] = *(const uint4*)(bp + kb8);
      bb[ns].q[1] = *(const uint4*)(bp + 16 + kb8);
    }
#pragma unroll
    for (int ms = 0; ms < 2; ++ms)
#pragma unroll
      for (int ns = 0; ns < 4; ++ns)
        acc[ms][ns] = wmma_bf16(a[ms].v, bb[ns].v, acc[ms][ns]);
  }

  // ---- epilogue ----
  int halfo = (lane >> 4) << 3;
  int cl = lane & 15;
#pragma unroll
  for (int ms = 0; ms < 2; ++ms)
#pragma unroll
    for (int ns = 0; ns < 4; ++ns)
#pragma unroll
      for (int r = 0; r < 8; ++r) {
        int row = m0 + wm * 32 + ms * 16 + halfo + r;
        int col = n0 + wn * 64 + ns * 16 + cl;
        if (row < M && col < N) {
          float v = acc[ms][ns][r];
          if (bias)  v += bias[col];
          if (gelu)  v = 0.5f * v * (1.f + erff(v * 0.70710678118654752f));
          if (resid) v += resid[(size_t)row * ldc + col];
          if (Cf)  Cf[(size_t)row * ldc + col] = v;
          if (Cbf) Cbf[(size_t)row * ldc + col] = f2bf(v);
        }
      }
}

// ---------------- flash attention (causal, hd=64), all bf16 I/O ----------
// grid: (B*H, S/64), block: 128 (4 waves, 16 q-rows each); double-buffered KV
__global__ __launch_bounds__(128) void k_attn(
    const bf16_t* __restrict__ Qb,   // qkv buffer, q at col 0, ld = 3*Dm
    const bf16_t* __restrict__ Kf,   // [B*S, Dm]
    const bf16_t* __restrict__ Vf,   // [B*S, Dm]
    bf16_t* __restrict__ O) {        // [B*S, Dm]
  __shared__ __attribute__((aligned(16))) unsigned short Qs[64 * 72];
  __shared__ __attribute__((aligned(16))) unsigned short Ks[2][32 * 72];
  __shared__ __attribute__((aligned(16))) unsigned short Vt[2][64 * LSTR]; // [hd][key]
  __shared__ __attribute__((aligned(16))) unsigned short Ps[4 * 16 * LSTR];
  unsigned qsb = (unsigned)(size_t)(void*)Qs;
  unsigned ksb = (unsigned)(size_t)(void*)Ks;

  int bh = blockIdx.x;
  int b = bh / Hh, h = bh % Hh;
  int qbase = blockIdx.y * 64;
  int tid = threadIdx.x, lane = tid & 31, wv = tid >> 5;
  size_t rowoff = (size_t)b * Sq;
  const float scale = 0.125f;   // 1/sqrt(64)

  // stage Q tile (64 rows x 8 chunks of 8 halves) via async copy
#pragma unroll
  for (int i = 0; i < 4; ++i) {
    int idx = tid + i * 128;
    int r = idx >> 3, c = idx & 7;
    async_ld_b128(qsb + r * 144 + c * 16,
                  Qb + (rowoff + qbase + r) * (3 * Dm) + h * HDm + c * 8);
  }
  wait_async();
  __syncthreads();

  int m16 = lane & 15;
  int kb8 = (lane >> 4) << 3;
  int b16 = (lane >> 4) << 4;
  int halfo = (lane >> 4) << 3;
  int cl = lane & 15;

  FragBF qa[2];
  {
    const unsigned short* qp = &Qs[(wv * 16 + m16) * 72];
    qa[0].q[0] = *(const uint4*)(qp + kb8);
    qa[0].q[1] = *(const uint4*)(qp + 16 + kb8);
    qa[1].q[0] = *(const uint4*)(qp + 32 + kb8);
    qa[1].q[1] = *(const uint4*)(qp + 48 + kb8);
  }

  float mi[8], li[8];
  v8f o[4];
#pragma unroll
  for (int r = 0; r < 8; ++r) { mi[r] = -1e30f; li[r] = 0.f; }
#pragma unroll
  for (int t = 0; t < 4; ++t)
#pragma unroll
    for (int r = 0; r < 8; ++r) o[t][r] = 0.f;

  // stage one 32-key KV tile: K via async copy, V transposed manually
  auto stage_kv = [&](int j, int bufi) {
    unsigned kb_ = ksb + (unsigned)bufi * (32 * 72 * 2);
#pragma unroll
    for (int i = 0; i < 2; ++i) {
      int idx = tid + i * 128;
      int r = idx >> 3, c = idx & 7;
      async_ld_b128(kb_ + r * 144 + c * 16,
                    Kf + (rowoff + j + r) * Dm + h * HDm + c * 8);
    }
#pragma unroll
    for (int i = 0; i < 16; ++i) {
      int idx = tid + i * 128;
      int r = idx >> 6, c = idx & 63;
      Vt[bufi][c * LSTR + r] = Vf[(rowoff + j + r) * Dm + h * HDm + c];
    }
  };

  int njt = (qbase + 64) / 32;
  stage_kv(0, 0);
  for (int jt = 0; jt < njt; ++jt) {
    int j = jt * 32;
    wait_async();
    __syncthreads();
    if (jt + 1 < njt) stage_kv((jt + 1) * 32, (jt + 1) & 1);
    int bi = jt & 1;

    // S = Q K^T
    v8f s[2];
#pragma unroll
    for (int nt = 0; nt < 2; ++nt) {
#pragma unroll
      for (int r = 0; r < 8; ++r) s[nt][r] = 0.f;
      const unsigned short* kp = &Ks[bi][(nt * 16 + m16) * 72];
      FragBF kb0, kb1;
      kb0.q[0] = *(const uint4*)(kp + kb8);
      kb0.q[1] = *(const uint4*)(kp + 16 + kb8);
      kb1.q[0] = *(const uint4*)(kp + 32 + kb8);
      kb1.q[1] = *(const uint4*)(kp + 48 + kb8);
      s[nt] = wmma_bf16(qa[0].v, kb0.v, s[nt]);
      s[nt] = wmma_bf16(qa[1].v, kb1.v, s[nt]);
    }

    // online softmax (rows live across 16-lane halves)
#pragma unroll
    for (int r = 0; r < 8; ++r) {
      int qrow = qbase + wv * 16 + halfo + r;
#pragma unroll
      for (int nt = 0; nt < 2; ++nt) {
        int kcol = j + nt * 16 + cl;
        float v = s[nt][r] * scale;
        if (kcol > qrow) v = -1e30f;
        s[nt][r] = v;
      }
      float rm = fmaxf(s[0][r], s[1][r]);
#pragma unroll
      for (int off = 8; off >= 1; off >>= 1)
        rm = fmaxf(rm, __shfl_xor(rm, off, 32));
      float nmx = fmaxf(mi[r], rm);
      float corr = __expf(mi[r] - nmx);
      float p0 = __expf(s[0][r] - nmx);
      float p1 = __expf(s[1][r] - nmx);
      s[0][r] = p0; s[1][r] = p1;
      float rs = p0 + p1;
#pragma unroll
      for (int off = 8; off >= 1; off >>= 1)
        rs += __shfl_xor(rs, off, 32);
      li[r] = li[r] * corr + rs;
      mi[r] = nmx;
#pragma unroll
      for (int t = 0; t < 4; ++t) o[t][r] *= corr;
    }

    // P (C-layout) -> LDS -> reload as A-fragment (per-wave patch)
    unsigned short* pw = &Ps[wv * 16 * LSTR];
#pragma unroll
    for (int r = 0; r < 8; ++r) {
      pw[(halfo + r) * LSTR + cl]      = f2bf(s[0][r]);
      pw[(halfo + r) * LSTR + 16 + cl] = f2bf(s[1][r]);
    }
    asm volatile("s_wait_dscnt 0" ::: "memory");   // same-wave LDS RAW
    FragBF pa;
    pa.q[0] = *(const uint4*)(pw + m16 * LSTR + kb8);
    pa.q[1] = *(const uint4*)(pw + m16 * LSTR + 16 + kb8);

    // O += P @ V
#pragma unroll
    for (int t = 0; t < 4; ++t) {
      const unsigned short* vp = &Vt[bi][(t * 16 + m16) * LSTR];
      FragBF vb;
      vb.q[0] = *(const uint4*)(vp + kb8);
      vb.q[1] = *(const uint4*)(vp + 16 + kb8);
      o[t] = wmma_bf16(pa.v, vb.v, o[t]);
    }
  }

  // epilogue: O / l -> bf16
#pragma unroll
  for (int t = 0; t < 4; ++t)
#pragma unroll
    for (int r = 0; r < 8; ++r) {
      int qrow = qbase + wv * 16 + halfo + r;
      O[(rowoff + qrow) * Dm + h * HDm + t * 16 + cl] = f2bf(o[t][r] / li[r]);
    }
}

// ---------------- host orchestration ----------------
extern "C" void kernel_launch(void* const* d_in, const int* in_sizes, int n_in,
                              void* d_out, int out_size, void* d_ws, size_t ws_size,
                              hipStream_t stream) {
  (void)in_sizes; (void)n_in; (void)out_size; (void)ws_size;
  const int*   ids     = (const int*)  d_in[0];
  const float* wte     = (const float*)d_in[1];
  const float* wpe     = (const float*)d_in[2];
  const float* ln1_w   = (const float*)d_in[3];
  const float* ln1_b   = (const float*)d_in[4];
  const float* attn_w  = (const float*)d_in[5];
  const float* attn_b  = (const float*)d_in[6];
  const float* kproj   = (const float*)d_in[7];
  const float* kmean   = (const float*)d_in[8];
  const float* vproj   = (const float*)d_in[9];
  const float* vmean   = (const float*)d_in[10];
  const float* cproj_w = (const float*)d_in[11];
  const float* cproj_b = (const float*)d_in[12];
  const float* ln2_w   = (const float*)d_in[13];
  const float* ln2_b   = (const float*)d_in[14];
  const float* fc_w    = (const float*)d_in[15];
  const float* fc_b    = (const float*)d_in[16];
  const float* pr_w    = (const float*)d_in[17];
  const float* pr_b    = (const float*)d_in[18];
  const float* lnf_w   = (const float*)d_in[19];
  const float* lnf_b   = (const float*)d_in[20];
  const float* lm_head = (const float*)d_in[21];
  float* logits = (float*)d_out;

  // workspace partition (256B aligned carve)
  char* base = (char*)d_ws;
  size_t off = 0;
  auto carve = [&](size_t bytes) -> char* {
    char* p = base + off;
    off += (bytes + 255) & ~(size_t)255;
    return p;
  };
  float*  xa   = (float*) carve((size_t)Mrows * Dm * 4);
  float*  xb   = (float*) carve((size_t)Mrows * Dm * 4);
  bf16_t* nbuf = (bf16_t*)carve((size_t)Mrows * Dm * 2);
  bf16_t* qkv  = (bf16_t*)carve((size_t)Mrows * 3 * Dm * 2);
  bf16_t* kc   = (bf16_t*)carve((size_t)Mrows * Kq * 2);
  bf16_t* vc   = (bf16_t*)carve((size_t)Mrows * Kq * 2);
  bf16_t* kf   = (bf16_t*)carve((size_t)Mrows * Dm * 2);
  bf16_t* vf   = (bf16_t*)carve((size_t)Mrows * Dm * 2);
  bf16_t* ao   = (bf16_t*)carve((size_t)Mrows * Dm * 2);
  bf16_t* hbuf = (bf16_t*)carve((size_t)Mrows * 4 * Dm * 2);
  // per-layer bf16 weight arena (re-used each layer) + lm_head arena
  bf16_t* w_aw = (bf16_t*)carve((size_t)3 * Dm * Dm * 2);
  bf16_t* w_kp = (bf16_t*)carve((size_t)Kq * Dm * 2);
  bf16_t* w_vp = (bf16_t*)carve((size_t)Kq * Dm * 2);
  bf16_t* w_cw = (bf16_t*)carve((size_t)Dm * Dm * 2);
  bf16_t* w_fw = (bf16_t*)carve((size_t)4 * Dm * Dm * 2);
  bf16_t* w_pw = (bf16_t*)carve((size_t)4 * Dm * Dm * 2);
  bf16_t* w_lm = (bf16_t*)carve((size_t)Vq * Dm * 2);

  auto cvt = [&](const float* src, bf16_t* dst, long n) {
    long blocks = (n / 8 + 255) / 256;
    if (blocks > 4096) blocks = 4096;
    k_cvtbf<<<(int)blocks, 256, 0, stream>>>(src, dst, n);
  };
  auto gemm = [&](const bf16_t* A, int lda, const bf16_t* W, const float* bias,
                  const float* resid, const float* amean,
                  float* Cf, bf16_t* Cbf, int ldc,
                  int M_, int N_, int Kd_, int transb, int gelu) {
    dim3 g((N_ + BN - 1) / BN, (M_ + BM - 1) / BM);
    k_gemm<<<g, 256, 0, stream>>>(A, lda, W, bias, resid, amean, Cf, Cbf, ldc,
                                  M_, N_, Kd_, transb, gelu);
  };

  k_embed<<<Mrows, 256, 0, stream>>>(ids, wte, wpe, xa);

  for (int i = 0; i < Lq; ++i) {
    const float* ab = attn_b  + (size_t)i * 3 * Dm;
    const float* km = kmean   + (size_t)i * Dm;
    const float* vm = vmean   + (size_t)i * Dm;
    const float* cb = cproj_b + (size_t)i * Dm;
    const float* fb = fc_b    + (size_t)i * 4 * Dm;
    const float* pb = pr_b    + (size_t)i * Dm;

    // convert this layer's weights to bf16 once
    cvt(attn_w  + (size_t)i * 3 * Dm * Dm, w_aw, (long)3 * Dm * Dm);
    cvt(kproj   + (size_t)i * Kq * Dm,     w_kp, (long)Kq * Dm);
    cvt(vproj   + (size_t)i * Kq * Dm,     w_vp, (long)Kq * Dm);
    cvt(cproj_w + (size_t)i * Dm * Dm,     w_cw, (long)Dm * Dm);
    cvt(fc_w    + (size_t)i * 4 * Dm * Dm, w_fw, (long)4 * Dm * Dm);
    cvt(pr_w    + (size_t)i * Dm * 4 * Dm, w_pw, (long)4 * Dm * Dm);

    // --- attention block ---
    k_ln<<<Mrows, 256, 0, stream>>>(xa, ln1_w + (size_t)i * Dm,
                                    ln1_b + (size_t)i * Dm, nbuf);
    gemm(nbuf, Dm, w_aw, ab, nullptr, nullptr, nullptr, qkv, 3 * Dm,
         Mrows, 3 * Dm, Dm, 0, 0);
    // kc = (k - kmean) @ kproj^T ; vc likewise   (manual A path: mean subtract)
    gemm(qkv + Dm,     3 * Dm, w_kp, nullptr, nullptr, km, nullptr, kc, Kq,
         Mrows, Kq, Dm, 0, 0);
    gemm(qkv + 2 * Dm, 3 * Dm, w_vp, nullptr, nullptr, vm, nullptr, vc, Kq,
         Mrows, Kq, Dm, 0, 0);
    // kf = kc @ kproj + kmean (TRANSB) ; vf likewise
    gemm(kc, Kq, w_kp, km, nullptr, nullptr, nullptr, kf, Dm, Mrows, Dm, Kq, 1, 0);
    gemm(vc, Kq, w_vp, vm, nullptr, nullptr, nullptr, vf, Dm, Mrows, Dm, Kq, 1, 0);
    // causal flash attention
    k_attn<<<dim3(Bq * Hh, Sq / 64), 128, 0, stream>>>(qkv, kf, vf, ao);
    // x = x + ao @ cproj^T + cb   (f32 residual stream)
    gemm(ao, Dm, w_cw, cb, xa, nullptr, xb, nullptr, Dm, Mrows, Dm, Dm, 0, 0);

    // --- mlp block ---
    k_ln<<<Mrows, 256, 0, stream>>>(xb, ln2_w + (size_t)i * Dm,
                                    ln2_b + (size_t)i * Dm, nbuf);
    gemm(nbuf, Dm, w_fw, fb, nullptr, nullptr, nullptr, hbuf, 4 * Dm,
         Mrows, 4 * Dm, Dm, 0, 1);
    gemm(hbuf, 4 * Dm, w_pw, pb, xb, nullptr, xa, nullptr, Dm,
         Mrows, Dm, 4 * Dm, 0, 0);
  }

  k_ln<<<Mrows, 256, 0, stream>>>(xa, lnf_w, lnf_b, nbuf);
  cvt(lm_head, w_lm, (long)Vq * Dm);
  gemm(nbuf, Dm, w_lm, nullptr, nullptr, nullptr, logits, nullptr, Vq,
       Mrows, Vq, Dm, 0, 0);
}
